// DGI_77833397338524
// MI455X (gfx1250) — matrix operand order repaired and checked by reading, not compiled, not run on hardware
//
#include <hip/hip_runtime.h>
#include <hip/hip_bf16.h>
#include <math.h>

// ---------------------------------------------------------------------------
// Types for CDNA5 WMMA (gfx1250, wave32)
// ---------------------------------------------------------------------------
typedef __attribute__((ext_vector_type(16))) __bf16 v16bf;
typedef __attribute__((ext_vector_type(2)))  __bf16 v2bf;
typedef __attribute__((ext_vector_type(8)))  float  v8f;

union FragBF {
    v16bf        v;
    unsigned int u[8];   // u[i] = elements 2i (lo half) and 2i+1 (hi half)
};

// Pack one float pair -> packed bf16 dword (lowers to v_cvt_pk_bf16_f32).
__device__ __forceinline__ unsigned int pk2(float lo, float hi) {
#if __has_builtin(__builtin_amdgcn_cvt_pk_bf16_f32)
    union { v2bf b; unsigned int u; } r;
    r.b = __builtin_amdgcn_cvt_pk_bf16_f32(lo, hi);
    return r.u;
#else
    union { v2bf b; unsigned int u; } r;
    r.b[0] = (__bf16)lo;
    r.b[1] = (__bf16)hi;
    return r.u;
#endif
}

__device__ __forceinline__ void cvt16(FragBF& f, float4 x0, float4 x1,
                                      float4 x2, float4 x3) {
    f.u[0] = pk2(x0.x, x0.y); f.u[1] = pk2(x0.z, x0.w);
    f.u[2] = pk2(x1.x, x1.y); f.u[3] = pk2(x1.z, x1.w);
    f.u[4] = pk2(x2.x, x2.y); f.u[5] = pk2(x2.z, x2.w);
    f.u[6] = pk2(x3.x, x3.y); f.u[7] = pk2(x3.z, x3.w);
}

#define NH 256   // hidden width (output columns of both GEMMs)

// ---------------------------------------------------------------------------
// Weight pre-pack: fp32 -> packed bf16 dwords, row-major preserved.
// ---------------------------------------------------------------------------
__global__ __launch_bounds__(256) void pack_w_bf16(
    const float* __restrict__ src, unsigned int* __restrict__ dst, int nPairs)
{
    int i      = blockIdx.x * blockDim.x + threadIdx.x;
    int stride = gridDim.x * blockDim.x;
    for (; i < nPairs; i += stride) {
        float2 x = ((const float2*)src)[i];
        dst[i] = pk2(x.x, x.y);
    }
}

// ---------------------------------------------------------------------------
// Fused GEMM (K compile-time): Out[m,n] = relu( sum_k A[m,k]*W[n,k] + bias[n] )
//   A   : [M x K]    fp32 row-major, cvt_pk'd to bf16 in-register once
//   Wpk : [NH x K/2] packed-bf16 dwords, row-major (pre-packed weights)
// Wave owns 16 rows x 256 cols (16 f32 accumulators). All 16 W-tile
// addresses are ONE base pointer + compile-time immediate offsets.
// B is 1-deep software-pipelined; next K-step's A loads issue before the
// 16-WMMA chain so HBM latency hides under the matrix math.
// ---------------------------------------------------------------------------
template<int K>
__global__ __launch_bounds__(256) void gemm_bias_relu_wmma(
    const float* __restrict__ A, int M,
    const unsigned int* __restrict__ Wpk,
    const float* __restrict__ bias,
    float* __restrict__ Out)
{
    constexpr int Kd = K >> 1;         // W row length in dwords
    constexpr int WT = 16 * Kd;        // dword stride between N-tiles

    const int tid  = threadIdx.x;
    const int lane = tid & 31;
    const int w    = tid >> 5;         // wave 0..7 -> 16-row slice
    const int lh   = lane & 15;        // lane-in-half
    const int g    = lane >> 4;        // half-wave group

    const int mBase = blockIdx.x * 128 + w * 16;
    int mRow = mBase + lh;
    if (mRow >= M) mRow = M - 1;       // clamp loads; stores are guarded

    const float* __restrict__ arow = A + (size_t)mRow * K + 16 * g;

    // Lane's base into packed W: row lh, K-offset 16g (= 8g dwords)
    const unsigned int* __restrict__ wk = Wpk + (size_t)lh * Kd + 8 * g;

    v8f acc[16] = {};                  // 16 x (16x16 f32) accumulators

    // Preload first A block (K = 16g..16g+15 ; contiguous, matches packed W)
    float4 x0 = *(const float4*)(arow);
    float4 x1 = *(const float4*)(arow + 4);
    float4 x2 = *(const float4*)(arow + 8);
    float4 x3 = *(const float4*)(arow + 12);

    for (int kb = 0; kb < K; kb += 32, wk += 16) {
        __builtin_prefetch(arow + kb + 128, 0, 1);   // stream-in hint

        FragBF a;
        cvt16(a, x0, x1, x2, x3);

        // Issue next K-step's A loads now (branchless clamped address)
        {
            const int kn = (kb + 32 < K) ? (kb + 32) : kb;
            const float* an = arow + kn;
            x0 = *(const float4*)(an);
            x1 = *(const float4*)(an + 4);
            x2 = *(const float4*)(an + 8);
            x3 = *(const float4*)(an + 12);
        }

        // B tile 0 (immediate-offset b128 pair), then 1-deep pipeline
        uint4 cb0 = *(const uint4*)(wk);
        uint4 cb1 = *(const uint4*)(wk + 4);
        #pragma unroll
        for (int t = 0; t < 16; ++t) {
            uint4 nb0 = cb0, nb1 = cb1;
            if (t < 15) {
                nb0 = *(const uint4*)(wk + (t + 1) * WT);
                nb1 = *(const uint4*)(wk + (t + 1) * WT + 4);
            }
            FragBF b;
            b.u[0] = cb0.x; b.u[1] = cb0.y; b.u[2] = cb0.z; b.u[3] = cb0.w;
            b.u[4] = cb1.x; b.u[5] = cb1.y; b.u[6] = cb1.z; b.u[7] = cb1.w;
            acc[t] = __builtin_amdgcn_wmma_f32_16x16x32_bf16(
                         false, a.v, false, b.v, (short)0, acc[t], false, false);
            cb0 = nb0; cb1 = nb1;
        }
    }

    // ---- epilogue: bias + ReLU; C/D layout: M = r + 8g, N = lh ----
    #pragma unroll
    for (int t = 0; t < 16; ++t) {
        const int col  = 16 * t + lh;
        const float bc = bias[col];
        #pragma unroll
        for (int r = 0; r < 8; ++r) {
            const int mo = mBase + r + 8 * g;
            if (mo < M) {
                float v = acc[t][r] + bc;
                Out[(size_t)mo * NH + col] = v > 0.f ? v : 0.f;
            }
        }
    }
}

// ---------------------------------------------------------------------------
// SpMM scatter: Agg[rows[e], :] += vals[e] * H[cols[e], :]
// One wave per edge; lane owns 8 contiguous features (2 x float4 gather,
// 8 x global_atomic_add_f32 into L2-resident Agg).
// ---------------------------------------------------------------------------
__global__ __launch_bounds__(256) void spmm_scatter(
    const int* __restrict__ rows, const int* __restrict__ cols,
    const float* __restrict__ vals, const float* __restrict__ H,
    float* __restrict__ Agg, int nEdges)
{
    const int lane = threadIdx.x & 31;
    const int wpb  = blockDim.x >> 5;
    const int wid  = blockIdx.x * wpb + (threadIdx.x >> 5);
    const int nW   = gridDim.x * wpb;
    for (int e = wid; e < nEdges; e += nW) {
        const int   r = rows[e];
        const int   c = cols[e];
        const float v = vals[e];
        const float4* hp = (const float4*)(H + (size_t)c * NH) + lane * 2;
        float4 h0 = hp[0], h1 = hp[1];
        float* ap = Agg + (size_t)r * NH + lane * 8;
        atomicAdd(ap + 0, v * h0.x); atomicAdd(ap + 1, v * h0.y);
        atomicAdd(ap + 2, v * h0.z); atomicAdd(ap + 3, v * h0.w);
        atomicAdd(ap + 4, v * h1.x); atomicAdd(ap + 5, v * h1.y);
        atomicAdd(ap + 6, v * h1.z); atomicAdd(ap + 7, v * h1.w);
    }
}

// ---------------------------------------------------------------------------
// Utility: zero a float buffer.
// ---------------------------------------------------------------------------
__global__ void zero_f32(float* __restrict__ p, size_t n)
{
    size_t i      = (size_t)blockIdx.x * blockDim.x + threadIdx.x;
    size_t stride = (size_t)gridDim.x * blockDim.x;
    for (; i < n; i += stride) p[i] = 0.f;
}

// ---------------------------------------------------------------------------
// Column partial sums of H (for mean over nodes). Thread t owns column t.
// ---------------------------------------------------------------------------
__global__ __launch_bounds__(256) void col_sum_partial(
    const float* __restrict__ H, int M, float* __restrict__ csum)
{
    const int t  = threadIdx.x;
    const int r0 = blockIdx.x * 256;
    int r1 = r0 + 256; if (r1 > M) r1 = M;
    float s = 0.f;
    for (int r = r0; r < r1; ++r) s += H[(size_t)r * NH + t];
    atomicAdd(&csum[t], s);
}

// ---------------------------------------------------------------------------
// c = sigmoid(csum / M); cw[j] = sum_h c[h] * Wb[h,j]
// ---------------------------------------------------------------------------
__global__ __launch_bounds__(256) void sigmoid_cw(
    const float* __restrict__ csum, const float* __restrict__ Wb,
    float* __restrict__ cw, int M)
{
    __shared__ float c[NH];
    const int t = threadIdx.x;
    c[t] = 1.f / (1.f + expf(-(csum[t] / (float)M)));
    __syncthreads();
    float s = 0.f;
    for (int h = 0; h < NH; ++h) s += c[h] * Wb[h * NH + t];
    cw[t] = s;
}

// ---------------------------------------------------------------------------
// out[s] = dot(H[nodes[s], :], cw) + bil_b   (wave-per-sample, shfl reduce)
// ---------------------------------------------------------------------------
__global__ __launch_bounds__(256) void bilinear_score(
    const float* __restrict__ H, const int* __restrict__ nodes,
    const float* __restrict__ cw, const float* __restrict__ bil_b,
    float* __restrict__ out, int nSamp)
{
    const int lane = threadIdx.x & 31;
    const int s    = blockIdx.x * (blockDim.x >> 5) + (threadIdx.x >> 5);
    if (s >= nSamp) return;
    const int node = nodes[s];
    const float4* hp = (const float4*)(H + (size_t)node * NH) + lane * 2;
    const float4* cp = (const float4*)cw + lane * 2;
    float4 h0 = hp[0], h1 = hp[1], c0 = cp[0], c1 = cp[1];
    float d = h0.x * c0.x + h0.y * c0.y + h0.z * c0.z + h0.w * c0.w
            + h1.x * c1.x + h1.y * c1.y + h1.z * c1.z + h1.w * c1.w;
    #pragma unroll
    for (int off = 16; off > 0; off >>= 1) d += __shfl_xor(d, off, 32);
    if (lane == 0) out[s] = d + bil_b[0];
}

// ---------------------------------------------------------------------------
// Orchestration
// ---------------------------------------------------------------------------
extern "C" void kernel_launch(void* const* d_in, const int* in_sizes, int n_in,
                              void* d_out, int out_size, void* d_ws, size_t ws_size,
                              hipStream_t stream)
{
    (void)n_in; (void)out_size; (void)ws_size;

    const int*   edge_rows = (const int*)  d_in[0];
    const int*   edge_cols = (const int*)  d_in[1];
    const float* edge_vals = (const float*)d_in[2];
    const float* AX1       = (const float*)d_in[3];
    const float* AX2       = (const float*)d_in[4];
    const int*   nodes     = (const int*)  d_in[5];
    const float* Wr_w      = (const float*)d_in[6];
    const float* Wr_b      = (const float*)d_in[7];
    const float* W_w       = (const float*)d_in[8];
    const float* W_b       = (const float*)d_in[9];
    const float* bil_W     = (const float*)d_in[10];
    const float* bil_b     = (const float*)d_in[11];
    float*       out       = (float*)d_out;

    const int nE = in_sizes[0];                 // 3,200,000
    const int nS = in_sizes[5];                 // 8,192
    const int K1 = 512;                         // N_IN
    const int K2 = 256;                         // N_H
    const int M  = in_sizes[3] / K1;            // 100,000

    // Workspace: hbuf | aggbuf | csum[256] | cw[256] | WrPk | WPk
    const size_t hElems = (size_t)M * NH;       // 25.6M floats = 102.4 MB
    float* hbuf   = (float*)d_ws;
    float* aggbuf = hbuf + hElems;
    float* csum   = aggbuf + hElems;
    float* cw     = csum + NH;
    unsigned int* WrPk = (unsigned int*)(cw + NH);          // NH*K1/2 dwords
    unsigned int* WPk  = WrPk + (size_t)NH * K1 / 2;        // NH*K2/2 dwords

    const dim3 blk(256);
    const int gGemm  = (M + 127) / 128;         // 128 rows / block
    const int gSpmm  = 2048;                    // 16K waves grid-striding edges
    const int gScore = (nS + 7) / 8;            // 8 waves / block

    // Pre-pack weights to bf16 (tiny; once per launch, deterministic)
    pack_w_bf16<<<128, blk, 0, stream>>>(Wr_w, WrPk, NH * K1 / 2);
    pack_w_bf16<<<64,  blk, 0, stream>>>(W_w,  WPk,  NH * K2 / 2);

    for (int graph = 0; graph < 2; ++graph) {
        const float* AX = (graph == 0) ? AX1 : AX2;

        // h = relu(AX @ Wr^T + Wr_b)            (bf16 WMMA, fp32 accum)
        gemm_bias_relu_wmma<512><<<gGemm, blk, 0, stream>>>(AX, M, WrPk, Wr_b, hbuf);

        // agg = SpMM(edges, h)
        zero_f32<<<2048, 256, 0, stream>>>(aggbuf, hElems);
        spmm_scatter<<<gSpmm, blk, 0, stream>>>(edge_rows, edge_cols, edge_vals,
                                                hbuf, aggbuf, nE);

        // h_final = relu(agg @ W^T + W_b)       (overwrites hbuf)
        gemm_bias_relu_wmma<256><<<gGemm, blk, 0, stream>>>(aggbuf, M, WPk, W_b, hbuf);

        if (graph == 0) {
            // c = sigmoid(mean_nodes(h_1));  cw = c @ bil_W
            zero_f32<<<1, 256, 0, stream>>>(csum, NH);
            col_sum_partial<<<(M + 255) / 256, blk, 0, stream>>>(hbuf, M, csum);
            sigmoid_cw<<<1, blk, 0, stream>>>(csum, bil_W, cw, M);
        }

        // sc_graph[s] = <h_final[nodes[s]], cw> + bil_b
        bilinear_score<<<gScore, blk, 0, stream>>>(hbuf, nodes, cw, bil_b,
                                                   out + (size_t)graph * nS, nS);
    }
}
